// DistMultDecoder_37718402793703
// MI455X (gfx1250) — matrix operand order repaired and checked by reading, not compiled, not run on hardware
//
#include <hip/hip_runtime.h>

// ---------------------------------------------------------------------------
// DistMult decoder: out = (input1 * w_r) @ input2^T + bias
// n1 = n2 = 8192, d = 512, fp32 in/out.
// Strategy: split-bf16 (bf16x3) GEMM on v_wmma_f32_16x16x32_bf16 (wave32),
// on-the-fly w_r scaling + hi/lo split during global->LDS staging,
// 128x128 block tile, K=32 LDS stages, double buffered.
// ---------------------------------------------------------------------------

typedef __attribute__((ext_vector_type(16))) __bf16       v16bf;
typedef __attribute__((ext_vector_type(4)))  __bf16       v4bf;
typedef __attribute__((ext_vector_type(8)))  float        v8f;
typedef __attribute__((ext_vector_type(4)))  float        v4f;
typedef __attribute__((ext_vector_type(4)))  unsigned int v4u;

#define BLK_M 128
#define BLK_N 128
#define BLK_K 32
#define LDSS  40   // 32 + 8 bf16 padding -> 80B row stride, conflict-free b128
#define NBUF  2

__global__ __launch_bounds__(256)
void distmult_wmma_bf16x3(const float* __restrict__ A,     // [n1][d]
                          const float* __restrict__ B,     // [n2][d]
                          const float* __restrict__ W,     // [ntypes][d]
                          const float* __restrict__ biasPtr,
                          const int*   __restrict__ typeIdx,
                          float*       __restrict__ C,     // [n1][n2]
                          int n1, int n2, int d)
{
    __shared__ __bf16 sAhi[NBUF][BLK_M * LDSS];
    __shared__ __bf16 sAlo[NBUF][BLK_M * LDSS];
    __shared__ __bf16 sBhi[NBUF][BLK_M * LDSS];
    __shared__ __bf16 sBlo[NBUF][BLK_M * LDSS];

    const int t    = threadIdx.x;
    const int lane = t & 31;
    const int wave = t >> 5;          // 0..7
    const int wM   = wave >> 2;       // 0..1  (64-row slab)
    const int wN   = wave & 3;        // 0..3  (32-col slab)
    const int l16  = lane & 15;
    const int hl   = lane >> 4;       // half-wave select

    const int bM = blockIdx.y * BLK_M;
    const int bN = blockIdx.x * BLK_N;

    const int ti = typeIdx[0];
    const float* __restrict__ wr = W + (size_t)ti * d;

    const int nSteps = d / BLK_K;     // 16

    const v8f vzero = {0.f, 0.f, 0.f, 0.f, 0.f, 0.f, 0.f, 0.f};
    v8f acc[4][2];
#pragma unroll
    for (int i = 0; i < 4; ++i)
#pragma unroll
        for (int j = 0; j < 2; ++j)
            acc[i][j] = vzero;

    // -------- staging helpers ------------------------------------------------
    v4f ra[4], rb[4];

    auto loadStep = [&](int step) {
        const int kk = step * BLK_K;
#pragma unroll
        for (int c = 0; c < 4; ++c) {
            const int chunk = t + 256 * c;        // 0..1023
            const int row   = chunk >> 3;         // 0..127
            const int col   = (chunk & 7) << 2;   // 0,4,...,28
            v4f wv = *(const v4f*)(wr + kk + col);
            v4f av = *(const v4f*)(A + (size_t)(bM + row) * d + kk + col);
            ra[c]  = av * wv;                      // fold w_r scale into A
            rb[c]  = *(const v4f*)(B + (size_t)(bN + row) * d + kk + col);
        }
    };

    auto storeStage = [&](int buf) {
#pragma unroll
        for (int c = 0; c < 4; ++c) {
            const int chunk = t + 256 * c;
            const int row   = chunk >> 3;
            const int col   = (chunk & 7) << 2;
            const int off   = row * LDSS + col;
            v4bf ah, al, bh, bl;
#pragma unroll
            for (int e = 0; e < 4; ++e) {
                float xa = ra[c][e];
                __bf16 h = (__bf16)xa;
                ah[e] = h;
                al[e] = (__bf16)(xa - (float)h);   // residual -> ~16 mantissa bits
                float xb = rb[c][e];
                __bf16 hb = (__bf16)xb;
                bh[e] = hb;
                bl[e] = (__bf16)(xb - (float)hb);
            }
            *(v4bf*)(sAhi[buf] + off) = ah;
            *(v4bf*)(sAlo[buf] + off) = al;
            *(v4bf*)(sBhi[buf] + off) = bh;
            *(v4bf*)(sBlo[buf] + off) = bl;
        }
    };

    // ISA 16-bit A 16x32 layout: lane half h: K in [8h,8h+8) U [16+8h,16+8h+8)
    auto ldfragA = [&](const __bf16* base, int row) -> v16bf {
        union { v16bf v; v4u q[2]; } u;
        const __bf16* p = base + row * LDSS + hl * 8;
        u.q[0] = *(const v4u*)(p);
        u.q[1] = *(const v4u*)(p + 16);
        return u.v;
    };
    // ISA 16-bit B 32x16 layout: lane half h: K in [8h,8h+8) U [16+8h,16+8h+8)
    auto ldfragB = [&](const __bf16* base, int row) -> v16bf {
        union { v16bf v; v4u q[2]; } u;
        const __bf16* p = base + row * LDSS + hl * 8;
        u.q[0] = *(const v4u*)(p);
        u.q[1] = *(const v4u*)(p + 16);
        return u.v;
    };

    auto compute = [&](int buf) {
        v16bf ah[4], al[4];
#pragma unroll
        for (int i = 0; i < 4; ++i) {
            const int row = wM * 64 + i * 16 + l16;
            ah[i] = ldfragA(sAhi[buf], row);
            al[i] = ldfragA(sAlo[buf], row);
        }
        v16bf bh[2], bl[2];
#pragma unroll
        for (int j = 0; j < 2; ++j) {
            const int row = wN * 32 + j * 16 + l16;
            bh[j] = ldfragB(sBhi[buf], row);
            bl[j] = ldfragB(sBlo[buf], row);
        }
#pragma unroll
        for (int i = 0; i < 4; ++i)
#pragma unroll
            for (int j = 0; j < 2; ++j) {
                acc[i][j] = __builtin_amdgcn_wmma_f32_16x16x32_bf16(
                    false, ah[i], false, bh[j], (short)0, acc[i][j], false, false);
                acc[i][j] = __builtin_amdgcn_wmma_f32_16x16x32_bf16(
                    false, ah[i], false, bl[j], (short)0, acc[i][j], false, false);
                acc[i][j] = __builtin_amdgcn_wmma_f32_16x16x32_bf16(
                    false, al[i], false, bh[j], (short)0, acc[i][j], false, false);
            }
    };

    // -------- main pipeline: double-buffered K loop --------------------------
    loadStep(0);
    int cur = 0;
    for (int step = 0; step < nSteps; ++step) {
        storeStage(cur);
        __syncthreads();
        if (step + 1 < nSteps) loadStep(step + 1);  // global latency hidden by WMMA
        compute(cur);
        cur ^= 1;
    }

    // -------- epilogue: bias add + coalesced f32 stores ----------------------
    const float bv = biasPtr[0];
#pragma unroll
    for (int i = 0; i < 4; ++i) {
#pragma unroll
        for (int j = 0; j < 2; ++j) {
            union { v8f v; float f[8]; } u;
            u.v = acc[i][j];
            const int col = bN + wN * 32 + j * 16 + l16;
#pragma unroll
            for (int r = 0; r < 8; ++r) {
                const int row = bM + wM * 64 + i * 16 + hl * 8 + r;
                C[(size_t)row * n2 + col] = u.f[r] + bv;
            }
        }
    }
}

extern "C" void kernel_launch(void* const* d_in, const int* in_sizes, int n_in,
                              void* d_out, int out_size, void* d_ws, size_t ws_size,
                              hipStream_t stream) {
    (void)n_in; (void)out_size; (void)d_ws; (void)ws_size;
    const float* A    = (const float*)d_in[0];   // input1 [n1][d]
    const float* B    = (const float*)d_in[1];   // input2 [n2][d]
    const float* W    = (const float*)d_in[2];   // weight [64][d]
    const float* bias = (const float*)d_in[3];   // [1]
    const int*   ti   = (const int*)d_in[4];     // scalar type_index
    float* C = (float*)d_out;

    const int d  = 512;
    const int n1 = in_sizes[0] / d;
    const int n2 = in_sizes[1] / d;

    dim3 grid(n2 / BLK_N, n1 / BLK_M);
    distmult_wmma_bf16x3<<<grid, 256, 0, stream>>>(A, B, W, bias, ti, C, n1, n2, d);
}
